// SingleLayerLstm_82652350644620
// MI455X (gfx1250) — matrix operand order repaired and checked by reading, not compiled, not run on hardware
//
#include <hip/hip_runtime.h>
#include <hip/hip_bf16.h>
#include <math.h>

// ---------------------------------------------------------------------------
// Problem constants (from reference): T=512, B=32, N_IN=1024, D=1024
// ---------------------------------------------------------------------------
#define TT   512
#define BB   32
#define NIN  1024
#define DD   1024
#define N4   4096        // 4*D
#define MM   (TT * BB)   // 16384 rows of the front GEMM

typedef __attribute__((ext_vector_type(16))) __bf16 v16bf;
typedef __attribute__((ext_vector_type(8)))  __bf16 v8bf;
typedef __attribute__((ext_vector_type(8)))  float  v8f;

union ABfrag { v16bf v; v8bf h[2]; };

// ---------------------------------------------------------------------------
// gfx1250 async LDS copy. Probe round 2 confirmed the builtin exists with
// signature (addrspace(1) int4*, addrspace(3) int4*, imm, imm).
// ---------------------------------------------------------------------------
#if defined(__has_builtin)
#  if __has_builtin(__builtin_amdgcn_global_load_async_to_lds_b128)
#    define HAVE_ASYNC_LDS 1
#  else
#    define HAVE_ASYNC_LDS 0
#  endif
#else
#  define HAVE_ASYNC_LDS 0
#endif

typedef __attribute__((__vector_size__(4 * sizeof(int)))) int int4v;
typedef __attribute__((address_space(1))) int4v as1_int4;
typedef __attribute__((address_space(3))) int4v as3_int4;

__device__ __forceinline__ void copy16_to_lds(const __bf16* g, __bf16* l) {
#if HAVE_ASYNC_LDS
    __builtin_amdgcn_global_load_async_to_lds_b128(
        (as1_int4*)g, (as3_int4*)l, 0, 0);
#else
    *reinterpret_cast<v8bf*>(l) = *reinterpret_cast<const v8bf*>(g);
#endif
}

__device__ __forceinline__ void async_wait_all() {
#if HAVE_ASYNC_LDS
#  if __has_builtin(__builtin_amdgcn_s_wait_asynccnt)
    __builtin_amdgcn_s_wait_asynccnt(0);
#  else
    asm volatile("s_wait_asynccnt 0x0" ::: "memory");
#  endif
#endif
}

// ---------------------------------------------------------------------------
// Prep kernels
// ---------------------------------------------------------------------------
__global__ __launch_bounds__(256) void f32_to_bf16_kernel(const float* __restrict__ in,
                                                          __bf16* __restrict__ out, int n) {
    int i = blockIdx.x * 256 + threadIdx.x;
    if (i < n) out[i] = (__bf16)in[i];
}

// out[c][r] (bf16) = in[r][c] (f32); R, C multiples of 32.
__global__ __launch_bounds__(256) void transpose_to_bf16_kernel(
        const float* __restrict__ in, __bf16* __restrict__ out, int R, int C) {
    __shared__ float tile[32][33];
    int bx = blockIdx.x * 32;          // column base (C)
    int by = blockIdx.y * 32;          // row base (R)
    int tx = threadIdx.x & 31;
    int ty = threadIdx.x >> 5;         // 0..7
#pragma unroll
    for (int p = 0; p < 4; ++p) {
        int r = by + ty + p * 8;
        tile[ty + p * 8][tx] = in[(size_t)r * C + bx + tx];
    }
    __syncthreads();
#pragma unroll
    for (int p = 0; p < 4; ++p) {
        int c = bx + ty + p * 8;
        out[(size_t)c * R + by + tx] = (__bf16)tile[tx][ty + p * 8];
    }
}

__global__ __launch_bounds__(256) void init_state_kernel(const float* __restrict__ h0,
                                                         const float* __restrict__ c0,
                                                         __bf16* __restrict__ hbf,
                                                         float* __restrict__ cst, int n) {
    int i = blockIdx.x * 256 + threadIdx.x;
    if (i < n) { hbf[i] = (__bf16)h0[i]; cst[i] = c0[i]; }
}

__global__ __launch_bounds__(256) void copy_f32_kernel(const float* __restrict__ in,
                                                       float* __restrict__ out, int n) {
    int i = blockIdx.x * 256 + threadIdx.x;
    if (i < n) out[i] = in[i];
}

// ---------------------------------------------------------------------------
// GEMM1: intern[TB x 4096] = Xb[TB x 1024] @ Wft^T + bf   (Wft is [4096][1024])
// Block tile 128(M) x 64(N), K-chunks of 32, bf16 WMMA, f32 accumulate.
// Tiles are DMA'd (async-to-LDS) in fragment-ready layout; no in-kernel
// transpose or conversion.
// ---------------------------------------------------------------------------
#define G1_MB 128
#define G1_NB 64
#define KC    32
#define LDP   40   // padded LDS stride in bf16 elements (80B, 16B-aligned)

__global__ __launch_bounds__(256) void gemm1_kernel(const __bf16* __restrict__ Xb,
                                                    const __bf16* __restrict__ Wft,
                                                    const float* __restrict__ bf,
                                                    float* __restrict__ intern) {
    __shared__ __bf16 As[G1_MB * LDP];   // [m][k]
    __shared__ __bf16 Bs[G1_NB * LDP];   // [n][k]

    const int tid  = threadIdx.x;
    const int lane = tid & 31;
    const int wave = tid >> 5;
    const int wm   = wave & 3;   // 0..3 : 32-row slab
    const int wn   = wave >> 2;  // 0..1 : 32-col slab
    const int m0   = blockIdx.y * G1_MB;
    const int n0   = blockIdx.x * G1_NB;
    const int hi   = lane >> 4;
    const int l15  = lane & 15;

    v8f acc[2][2] = {};

    for (int k0 = 0; k0 < NIN; k0 += KC) {
        __syncthreads();
        // ---- A tile: 128 rows x 32 k (bf16) = 512 chunks of 16B ----
#pragma unroll
        for (int p = 0; p < 2; ++p) {
            int q   = tid + p * 256;
            int row = q >> 2;
            int c   = q & 3;
            copy16_to_lds(Xb + (size_t)(m0 + row) * NIN + k0 + c * 8,
                          &As[row * LDP + c * 8]);
        }
        // ---- B tile: 64 rows (n) x 32 k (bf16) = 256 chunks of 16B ----
        {
            int row = tid >> 2;
            int c   = tid & 3;
            copy16_to_lds(Wft + (size_t)(n0 + row) * NIN + k0 + c * 8,
                          &Bs[row * LDP + c * 8]);
        }
        async_wait_all();
        __syncthreads();

        // ---- fragments (ISA VGPR layouts) + WMMA ----
        ABfrag af[2], bfg[2];
#pragma unroll
        for (int mi = 0; mi < 2; ++mi) {
            // A 16x32: j0..7 -> K=hi*8+j ; j8..15 -> K=16+hi*8+(j-8)
            const __bf16* base = &As[(wm * 32 + mi * 16 + l15) * LDP + hi * 8];
            af[mi].h[0] = *reinterpret_cast<const v8bf*>(base);
            af[mi].h[1] = *reinterpret_cast<const v8bf*>(base + 16);
        }
#pragma unroll
        for (int ni = 0; ni < 2; ++ni) {
            // B 32x16: j0..7 -> K=hi*8+j ; j8..15 -> K=16+hi*8+(j-8)
            const __bf16* base = &Bs[(wn * 32 + ni * 16 + l15) * LDP + hi * 8];
            bfg[ni].h[0] = *reinterpret_cast<const v8bf*>(base);
            bfg[ni].h[1] = *reinterpret_cast<const v8bf*>(base + 16);
        }
#pragma unroll
        for (int mi = 0; mi < 2; ++mi)
#pragma unroll
            for (int ni = 0; ni < 2; ++ni)
                acc[mi][ni] = __builtin_amdgcn_wmma_f32_16x16x32_bf16(
                    false, af[mi].v, false, bfg[ni].v,
                    (short)0, acc[mi][ni], false, false);
    }

    // ---- epilogue: add bias, store f32 ----
#pragma unroll
    for (int mi = 0; mi < 2; ++mi)
#pragma unroll
        for (int ni = 0; ni < 2; ++ni) {
            int col = n0 + wn * 32 + ni * 16 + l15;
            float bias = bf[col];
#pragma unroll
            for (int e = 0; e < 8; ++e) {
                int row = m0 + wm * 32 + mi * 16 + hi * 8 + e;
                intern[(size_t)row * N4 + col] = acc[mi][ni][e] + bias;
            }
        }
}

// ---------------------------------------------------------------------------
// Fused LSTM step, one launch per t. Each block owns 16 d-columns; the 8
// waves are (gate 0..3) x (mi 0..1). With Wrt pre-transposed to [n][k] bf16,
// both fragments come straight from global/L2 — the K-loop has no LDS and no
// barriers. Gates are exchanged via LDS once, then the cell update runs.
// ---------------------------------------------------------------------------
__global__ __launch_bounds__(256) void lstm_step_kernel(int t,
        const __bf16* __restrict__ hprev,   // [32 x 1024] bf16 row-major
        const __bf16* __restrict__ Wrt,     // [4096 x 1024] bf16 (n-major)
        const float*  __restrict__ intern,  // [TB x 4096]
        const float*  __restrict__ Igate,   // [T x B]
        float*  __restrict__ cstate,        // [32 x 1024]
        __bf16* __restrict__ hnext,         // [32 x 1024] bf16
        float*  __restrict__ Yout,          // [T x B x D]
        float*  __restrict__ Cout) {        // [T x B x D]
    __shared__ float gs[4][32][16];         // gate pre-activations

    const int tid  = threadIdx.x;
    const int lane = tid & 31;
    const int wave = tid >> 5;
    const int gate = wave >> 1;    // 0..3
    const int mi   = wave & 1;     // batch-row slab
    const int d0   = blockIdx.x * 16;
    const int hi   = lane >> 4;
    const int l15  = lane & 15;

    const __bf16* arow = hprev + (size_t)(mi * 16 + l15) * DD;
    const __bf16* brow = Wrt + (size_t)(gate * DD + d0 + l15) * DD;

    v8f acc = {};
#pragma unroll 4
    for (int k0 = 0; k0 < DD; k0 += KC) {
        ABfrag af, bfg;
        af.h[0]  = *reinterpret_cast<const v8bf*>(arow + k0 + hi * 8);
        af.h[1]  = *reinterpret_cast<const v8bf*>(arow + k0 + hi * 8 + 16);
        bfg.h[0] = *reinterpret_cast<const v8bf*>(brow + k0 + hi * 8);
        bfg.h[1] = *reinterpret_cast<const v8bf*>(brow + k0 + hi * 8 + 16);
        acc = __builtin_amdgcn_wmma_f32_16x16x32_bf16(
            false, af.v, false, bfg.v, (short)0, acc, false, false);
    }

    // ---- add intern[t] and publish the gate tile to LDS ----
    {
        int col = gate * DD + d0 + l15;
#pragma unroll
        for (int e = 0; e < 8; ++e) {
            int m = mi * 16 + hi * 8 + e;          // batch row 0..31
            gs[gate][m][l15] = acc[e] + intern[((size_t)t * BB + m) * N4 + col];
        }
    }
    __syncthreads();

    // ---- pointwise LSTM cell update: 32x16 = 512 outputs, 2 per thread ----
#pragma unroll
    for (int p = 0; p < 2; ++p) {
        int idx = tid + p * 256;
        int m = idx >> 4;
        int c = idx & 15;
        int d = d0 + c;
        float cellin = tanhf(gs[0][m][c]);
        float ing = 1.0f / (1.0f + __expf(-gs[1][m][c]));
        float fg  = 1.0f / (1.0f + __expf(-gs[2][m][c]));
        float og  = 1.0f / (1.0f + __expf(-gs[3][m][c]));
        float cp  = cstate[m * DD + d];
        float it  = Igate[t * BB + m];
        float cnew = it * (fg * cp + ing * cellin) + (1.0f - it) * cp;
        float y    = it * tanhf(cnew) * og;
        size_t o = ((size_t)t * BB + m) * DD + d;
        Cout[o] = cnew;
        Yout[o] = y;
        cstate[m * DD + d] = cnew;
        hnext[m * DD + d]  = (__bf16)y;
    }
}

// ---------------------------------------------------------------------------
// Host-side orchestration
// ---------------------------------------------------------------------------
extern "C" void kernel_launch(void* const* d_in, const int* in_sizes, int n_in,
                              void* d_out, int out_size, void* d_ws, size_t ws_size,
                              hipStream_t stream) {
    const float* X  = (const float*)d_in[0];   // (T,B,NIN)
    const float* I  = (const float*)d_in[1];   // (T,B)
    const float* Wf = (const float*)d_in[2];   // (NIN,4D)
    const float* Wr = (const float*)d_in[3];   // (D,4D)
    const float* bf = (const float*)d_in[4];   // (4D,)
    const float* h0 = (const float*)d_in[5];   // (B,D)
    const float* c0 = (const float*)d_in[6];   // (B,D)

    // Workspace layout (bytes)
    char* ws = (char*)d_ws;
    const size_t INTERN_BYTES = (size_t)MM * N4 * sizeof(float);      // 256 MB
    const size_t XB_BYTES     = (size_t)MM * NIN * sizeof(__bf16);    // 32 MB
    const size_t WT_BYTES     = (size_t)DD * N4 * sizeof(__bf16);     // 8 MB each
    const size_t H_BYTES      = (size_t)BB * DD * sizeof(__bf16);     // 64 KB
    float*  intern = (float*)(ws);
    __bf16* Xb     = (__bf16*)(ws + INTERN_BYTES);
    __bf16* Wft    = (__bf16*)(ws + INTERN_BYTES + XB_BYTES);
    __bf16* Wrt    = (__bf16*)(ws + INTERN_BYTES + XB_BYTES + WT_BYTES);
    __bf16* hbf0   = (__bf16*)(ws + INTERN_BYTES + XB_BYTES + 2 * WT_BYTES);
    __bf16* hbf1   = (__bf16*)(ws + INTERN_BYTES + XB_BYTES + 2 * WT_BYTES + H_BYTES);
    float*  cstate = (float*)(ws + INTERN_BYTES + XB_BYTES + 2 * WT_BYTES + 2 * H_BYTES);

    // Output layout: Y (T*B*D) | C (T*B*D) | c_last (B*D)
    float* Y  = (float*)d_out;
    float* C  = Y + (size_t)TT * BB * DD;
    float* cl = C + (size_t)TT * BB * DD;

    // 1) X -> bf16
    {
        int n = MM * NIN;
        f32_to_bf16_kernel<<<(n + 255) / 256, 256, 0, stream>>>(X, Xb, n);
    }
    // 2) Wf, Wr -> transposed bf16 [n][k] (one-time; Wrt then lives in L2)
    {
        dim3 grid(N4 / 32, NIN / 32);   // 128 x 32
        transpose_to_bf16_kernel<<<grid, 256, 0, stream>>>(Wf, Wft, NIN, N4);
        transpose_to_bf16_kernel<<<grid, 256, 0, stream>>>(Wr, Wrt, DD, N4);
    }
    // 3) init mutable state from h0/c0 (inputs stay untouched)
    {
        int n = BB * DD;
        init_state_kernel<<<(n + 255) / 256, 256, 0, stream>>>(h0, c0, hbf0, cstate, n);
    }
    // 4) front GEMM: intern = X @ Wf + bf
    {
        dim3 grid(N4 / G1_NB, MM / G1_MB);   // 64 x 128
        gemm1_kernel<<<grid, 256, 0, stream>>>(Xb, Wft, bf, intern);
    }
    // 5) sequential scan: 512 fused step kernels (ping-pong bf16 h state)
    for (int t = 0; t < TT; ++t) {
        __bf16* hin  = (t & 1) ? hbf1 : hbf0;
        __bf16* hout = (t & 1) ? hbf0 : hbf1;
        lstm_step_kernel<<<DD / 16, 256, 0, stream>>>(
            t, hin, Wrt, intern, I, cstate, hout, Y, C);
    }
    // 6) c_last
    {
        int n = BB * DD;
        copy_f32_kernel<<<(n + 255) / 256, 256, 0, stream>>>(cstate, cl, n);
    }
}
